// MixtureOfExpertsPolicy_8976481648647
// MI455X (gfx1250) — compile-verified, matrix operand order checked by reading, not run on hardware
//
#include <hip/hip_runtime.h>
#include <hip/hip_bf16.h>

// Problem dims
#define B_  4096
#define I_  512
#define L_  128
#define D_  512
#define E_  8
#define H_  512
#define HR_ 256
#define A_  32

typedef __attribute__((ext_vector_type(16))) __bf16 v16bf;
typedef __attribute__((ext_vector_type(8)))  __bf16 v8bf;
typedef __attribute__((ext_vector_type(8)))  float  v8f;
typedef unsigned int v4u __attribute__((ext_vector_type(4)));
typedef int          v8i __attribute__((ext_vector_type(8)));
typedef int          v4i __attribute__((ext_vector_type(4)));

#define BM  64
#define BN  64
#define BKC 64   // K-chunk per TDM transfer (two WMMA K-steps)

// Build a 16x32 bf16 WMMA fragment for this lane from an LDS row.
// CDNA5 16-bit A/B layout: lanes 0-15 hold K {0..7,16..23},
// lanes 16-31 hold K {8..15,24..31}; row index = lane & 15.
__device__ __forceinline__ v16bf load_frag(const __bf16* rowPtr, int kb) {
  v8bf lo = *(const v8bf*)(rowPtr + kb);
  v8bf hi = *(const v8bf*)(rowPtr + kb + 16);
  return __builtin_shufflevector(lo, hi, 0,1,2,3,4,5,6,7,8,9,10,11,12,13,14,15);
}

// Issue one TDM 2-D tile load: rows x tileK bf16 elements, row stride strideK
// (elements), from gaddr into LDS byte address ldsAddr. rowsBound limits valid
// rows from the tile start (beyond -> zero fill). Uniform args; wave-scalar op.
__device__ __forceinline__ void tdm_load_2d(const __bf16* gaddr, unsigned ldsAddr,
                                            unsigned rowsBound, unsigned tileRows,
                                            unsigned tileK, unsigned strideK)
{
  unsigned long long ga = (unsigned long long)(uintptr_t)gaddr;
  v4u g0;
  g0[0] = 1u;                                            // count=1, load descriptor
  g0[1] = ldsAddr;                                       // lds_addr (bytes)
  g0[2] = (unsigned)(ga & 0xFFFFFFFFu);                  // global_addr[31:0]
  g0[3] = (unsigned)((ga >> 32) & 0x1FFFFFFu) | (2u << 30); // addr[56:32] | type=2
  unsigned long long s0 = (unsigned long long)strideK;   // dim0 stride (elements)
  v8i g1;
  g1[0] = (int)(1u << 16);                               // data_size=1 (2 bytes)
  g1[1] = (int)((tileK & 0xFFFFu) << 16);                // tensor_dim0[15:0]
  g1[2] = (int)(((tileK >> 16) & 0xFFFFu) | ((rowsBound & 0xFFFFu) << 16));
  g1[3] = (int)(((rowsBound >> 16) & 0xFFFFu) | ((tileK & 0xFFFFu) << 16)); // tile_dim0
  g1[4] = (int)(tileRows & 0xFFFFu);                     // tile_dim1 ; tile_dim2=0
  g1[5] = (int)(s0 & 0xFFFFFFFFu);                       // tensor_dim0_stride lo
  g1[6] = (int)((s0 >> 32) & 0xFFFFu);                   // stride hi ; dim1 stride=0
  g1[7] = 0;
  v4i z4 = {0, 0, 0, 0};
  v8i z8 = {0, 0, 0, 0, 0, 0, 0, 0};
  __builtin_amdgcn_tensor_load_to_lds(g0, g1, z4, z4, z8, 0);
}

// C(MxN) = A(MxK bf16 row-major) x W(NxK bf16 row-major)^T + bias
// M % 64 == 0, K % 64 == 0; N arbitrary (TDM zero-fills OOB rows of W).
__global__ __launch_bounds__(128) void k_gemm_bf16(
    const __bf16* __restrict__ Am, const __bf16* __restrict__ Wm,
    const float* __restrict__ bias, float* __restrict__ outF32,
    __bf16* __restrict__ outBF16, int M, int N, int K)
{
  __shared__ __align__(16) __bf16 sA[2][BM][BKC];
  __shared__ __align__(16) __bf16 sB[2][BN][BKC];
  const int tid  = threadIdx.x;
  const int lane = tid & 31;
  const int wave = tid >> 5;
  const int wm = (wave >> 1) * 32;
  const int wn = (wave & 1) * 32;
  const int m0 = blockIdx.x * BM;
  const int n0 = blockIdx.y * BN;
  const int fr = lane & 15;
  const int kb = (lane < 16) ? 0 : 8;
  const int rowAdd = (lane < 16) ? 0 : 8;
  const int nChunks = K / BKC;
  const unsigned rowsW = (unsigned)(N - n0 > 0 ? N - n0 : 0);

  v8f acc[2][2] = {};

  if (wave == 0) {
    tdm_load_2d(Am + (size_t)m0 * K, (unsigned)(uintptr_t)&sA[0][0][0],
                (unsigned)(M - m0), BM, BKC, (unsigned)K);
    tdm_load_2d(Wm + (size_t)n0 * K, (unsigned)(uintptr_t)&sB[0][0][0],
                rowsW, BN, BKC, (unsigned)K);
  }

  for (int c = 0; c < nChunks; ++c) {
    const int buf = c & 1;
    if (wave == 0) {
      if (c + 1 < nChunks) {
        const int k1 = (c + 1) * BKC;
        tdm_load_2d(Am + (size_t)m0 * K + k1, (unsigned)(uintptr_t)&sA[buf ^ 1][0][0],
                    (unsigned)(M - m0), BM, BKC, (unsigned)K);
        tdm_load_2d(Wm + (size_t)n0 * K + k1, (unsigned)(uintptr_t)&sB[buf ^ 1][0][0],
                    rowsW, BN, BKC, (unsigned)K);
        __builtin_amdgcn_s_wait_tensorcnt(2);   // oldest chunk complete
      } else {
        __builtin_amdgcn_s_wait_tensorcnt(0);
      }
    }
    __syncthreads();                            // chunk c data visible to all waves
    for (int ks = 0; ks < 2; ++ks) {
      v16bf af0 = load_frag(&sA[buf][wm + fr][ks * 32],      kb);
      v16bf af1 = load_frag(&sA[buf][wm + 16 + fr][ks * 32], kb);
      v16bf bf0 = load_frag(&sB[buf][wn + fr][ks * 32],      kb);
      v16bf bf1 = load_frag(&sB[buf][wn + 16 + fr][ks * 32], kb);
      acc[0][0] = __builtin_amdgcn_wmma_f32_16x16x32_bf16(false, af0, false, bf0, (short)0, acc[0][0], false, false);
      acc[0][1] = __builtin_amdgcn_wmma_f32_16x16x32_bf16(false, af0, false, bf1, (short)0, acc[0][1], false, false);
      acc[1][0] = __builtin_amdgcn_wmma_f32_16x16x32_bf16(false, af1, false, bf0, (short)0, acc[1][0], false, false);
      acc[1][1] = __builtin_amdgcn_wmma_f32_16x16x32_bf16(false, af1, false, bf1, (short)0, acc[1][1], false, false);
    }
    __syncthreads();                            // all done reading buf before reuse
  }

  for (int mi = 0; mi < 2; ++mi)
    for (int ni = 0; ni < 2; ++ni)
      for (int v = 0; v < 8; ++v) {
        int m = m0 + wm + mi * 16 + rowAdd + v;
        int n = n0 + wn + ni * 16 + fr;
        if (n < N) {
          float val = acc[mi][ni][v] + (bias ? bias[n] : 0.0f);
          size_t o = (size_t)m * N + n;
          if (outF32)  outF32[o]  = val;
          if (outBF16) outBF16[o] = (__bf16)val;
        }
      }
}

// combined[b,d] += gate[b,e] * (h_new[b,:] . W_proj[e][d,:] + b_proj[e][d])
__global__ __launch_bounds__(128) void k_gemm_combine(
    const __bf16* __restrict__ Am, const __bf16* __restrict__ Wm,
    const float* __restrict__ bias, const float* __restrict__ gate, int e,
    float* __restrict__ outF32, int M, int N, int K)
{
  __shared__ __align__(16) __bf16 sA[2][BM][BKC];
  __shared__ __align__(16) __bf16 sB[2][BN][BKC];
  const int tid  = threadIdx.x;
  const int lane = tid & 31;
  const int wave = tid >> 5;
  const int wm = (wave >> 1) * 32;
  const int wn = (wave & 1) * 32;
  const int m0 = blockIdx.x * BM;
  const int n0 = blockIdx.y * BN;
  const int fr = lane & 15;
  const int kb = (lane < 16) ? 0 : 8;
  const int rowAdd = (lane < 16) ? 0 : 8;
  const int nChunks = K / BKC;
  const unsigned rowsW = (unsigned)(N - n0 > 0 ? N - n0 : 0);

  v8f acc[2][2] = {};

  if (wave == 0) {
    tdm_load_2d(Am + (size_t)m0 * K, (unsigned)(uintptr_t)&sA[0][0][0],
                (unsigned)(M - m0), BM, BKC, (unsigned)K);
    tdm_load_2d(Wm + (size_t)n0 * K, (unsigned)(uintptr_t)&sB[0][0][0],
                rowsW, BN, BKC, (unsigned)K);
  }

  for (int c = 0; c < nChunks; ++c) {
    const int buf = c & 1;
    if (wave == 0) {
      if (c + 1 < nChunks) {
        const int k1 = (c + 1) * BKC;
        tdm_load_2d(Am + (size_t)m0 * K + k1, (unsigned)(uintptr_t)&sA[buf ^ 1][0][0],
                    (unsigned)(M - m0), BM, BKC, (unsigned)K);
        tdm_load_2d(Wm + (size_t)n0 * K + k1, (unsigned)(uintptr_t)&sB[buf ^ 1][0][0],
                    rowsW, BN, BKC, (unsigned)K);
        __builtin_amdgcn_s_wait_tensorcnt(2);
      } else {
        __builtin_amdgcn_s_wait_tensorcnt(0);
      }
    }
    __syncthreads();
    for (int ks = 0; ks < 2; ++ks) {
      v16bf af0 = load_frag(&sA[buf][wm + fr][ks * 32],      kb);
      v16bf af1 = load_frag(&sA[buf][wm + 16 + fr][ks * 32], kb);
      v16bf bf0 = load_frag(&sB[buf][wn + fr][ks * 32],      kb);
      v16bf bf1 = load_frag(&sB[buf][wn + 16 + fr][ks * 32], kb);
      acc[0][0] = __builtin_amdgcn_wmma_f32_16x16x32_bf16(false, af0, false, bf0, (short)0, acc[0][0], false, false);
      acc[0][1] = __builtin_amdgcn_wmma_f32_16x16x32_bf16(false, af0, false, bf1, (short)0, acc[0][1], false, false);
      acc[1][0] = __builtin_amdgcn_wmma_f32_16x16x32_bf16(false, af1, false, bf0, (short)0, acc[1][0], false, false);
      acc[1][1] = __builtin_amdgcn_wmma_f32_16x16x32_bf16(false, af1, false, bf1, (short)0, acc[1][1], false, false);
    }
    __syncthreads();
  }

  for (int mi = 0; mi < 2; ++mi)
    for (int ni = 0; ni < 2; ++ni)
      for (int v = 0; v < 8; ++v) {
        int m = m0 + wm + mi * 16 + rowAdd + v;
        int n = n0 + wn + ni * 16 + fr;
        if (n < N) {
          float g = gate[(size_t)m * E_ + e];
          float val = (acc[mi][ni][v] + bias[n]) * g;
          outF32[(size_t)m * N + n] += val;
        }
      }
}

// GRU gate math: h' = (1-z)*tanh(xn + r*hn) + z*h ; r,z = sigmoid(sum)
__global__ void k_gru(const float* __restrict__ gx, const float* __restrict__ gh,
                      const float* __restrict__ h, float* __restrict__ h_out,
                      __bf16* __restrict__ h_out_bf, int Btot, int Hh)
{
  int idx = blockIdx.x * blockDim.x + threadIdx.x;
  if (idx >= Btot * Hh) return;
  int b = idx / Hh, j = idx - b * Hh;
  size_t base = (size_t)b * 3 * Hh;
  float xr = gx[base + j], xz = gx[base + Hh + j], xn = gx[base + 2 * Hh + j];
  float hr = gh[base + j], hz = gh[base + Hh + j], hn = gh[base + 2 * Hh + j];
  float r = 1.0f / (1.0f + __expf(-(xr + hr)));
  float z = 1.0f / (1.0f + __expf(-(xz + hz)));
  float nct = tanhf(xn + r * hn);
  float out = (1.0f - z) * nct + z * h[idx];
  h_out[idx] = out;
  if (h_out_bf) h_out_bf[idx] = (__bf16)out;
}

__global__ void k_fc_softmax(const float* __restrict__ hR, const float* __restrict__ Wfc,
                             const float* __restrict__ bfc, float* __restrict__ gate)
{
  int b = blockIdx.x * blockDim.x + threadIdx.x;
  if (b >= B_) return;
  float lg[E_];
  float mx = -1e30f;
  const float* hrow = hR + (size_t)b * HR_;
  for (int e = 0; e < E_; ++e) {
    float s = bfc[e];
    const float* wrow = Wfc + (size_t)e * HR_;
    for (int k = 0; k < HR_; ++k) s += hrow[k] * wrow[k];
    lg[e] = s; mx = fmaxf(mx, s);
  }
  float den = 0.0f;
  for (int e = 0; e < E_; ++e) { lg[e] = __expf(lg[e] - mx); den += lg[e]; }
  float inv = 1.0f / den;
  for (int e = 0; e < E_; ++e) gate[(size_t)b * E_ + e] = lg[e] * inv;
}

__global__ void k_concat_bf16(const float* __restrict__ x, const float* __restrict__ lang,
                              __bf16* __restrict__ out)
{
  int idx = blockIdx.x * blockDim.x + threadIdx.x;
  if (idx >= B_ * (I_ + L_)) return;
  int b = idx / (I_ + L_), c = idx - b * (I_ + L_);
  float v = (c < I_) ? x[(size_t)b * I_ + c] : lang[(size_t)b * L_ + (c - I_)];
  out[idx] = (__bf16)v;
}

__global__ void k_cvt(const float* __restrict__ s, __bf16* __restrict__ d, int n)
{
  int idx = blockIdx.x * blockDim.x + threadIdx.x;
  if (idx < n) d[idx] = (__bf16)s[idx];
}

__global__ void k_zero(float* __restrict__ p, int n)
{
  int idx = blockIdx.x * blockDim.x + threadIdx.x;
  if (idx < n) p[idx] = 0.0f;
}

extern "C" void kernel_launch(void* const* d_in, const int* in_sizes, int n_in,
                              void* d_out, int out_size, void* d_ws, size_t ws_size,
                              hipStream_t stream)
{
  (void)in_sizes; (void)n_in; (void)out_size; (void)ws_size;
  const float* x      = (const float*)d_in[0];
  const float* lang   = (const float*)d_in[1];
  const float* hR0    = (const float*)d_in[2];
  const float* hE0    = (const float*)d_in[3];
  const float* W_in   = (const float*)d_in[4];
  const float* b_in   = (const float*)d_in[5];
  const float* Wih_r  = (const float*)d_in[6];
  const float* Whh_r  = (const float*)d_in[7];
  const float* bih_r  = (const float*)d_in[8];
  const float* bhh_r  = (const float*)d_in[9];
  const float* W_fc   = (const float*)d_in[10];
  const float* b_fc   = (const float*)d_in[11];
  const float* Wih_e  = (const float*)d_in[12];
  const float* Whh_e  = (const float*)d_in[13];
  const float* bih_e  = (const float*)d_in[14];
  const float* bhh_e  = (const float*)d_in[15];
  const float* W_proj = (const float*)d_in[16];
  const float* b_proj = (const float*)d_in[17];
  const float* W_head = (const float*)d_in[18];
  const float* b_head = (const float*)d_in[19];

  float* out_logits = (float*)d_out;                       // B x A
  float* out_hR     = out_logits + (size_t)B_ * A_;        // B x HR
  float* out_hE     = out_hR + (size_t)B_ * HR_;           // E x B x H

  char* ws = (char*)d_ws;
  size_t off = 0;
  auto alloc = [&](size_t bytes) -> void* {
    void* p = ws + off; off = (off + bytes + 255) & ~(size_t)255; return p;
  };
  __bf16* xcat      = (__bf16*)alloc((size_t)B_ * (I_ + L_) * 2);
  __bf16* xp_bf     = (__bf16*)alloc((size_t)B_ * D_ * 2);
  __bf16* w_in_bf   = (__bf16*)alloc((size_t)D_ * (I_ + L_) * 2);
  __bf16* wih_r_bf  = (__bf16*)alloc((size_t)3 * HR_ * D_ * 2);
  __bf16* whh_r_bf  = (__bf16*)alloc((size_t)3 * HR_ * HR_ * 2);
  __bf16* hR_bf     = (__bf16*)alloc((size_t)B_ * HR_ * 2);
  float*  gxbuf     = (float*)alloc((size_t)B_ * 3 * H_ * 4);   // reused router + experts
  float*  ghbuf     = (float*)alloc((size_t)B_ * 3 * H_ * 4);
  float*  gate      = (float*)alloc((size_t)B_ * E_ * 4);
  __bf16* wih_e_bf  = (__bf16*)alloc((size_t)3 * H_ * D_ * 2);  // per-expert, reused
  __bf16* whh_e_bf  = (__bf16*)alloc((size_t)3 * H_ * H_ * 2);
  __bf16* wproj_bf  = (__bf16*)alloc((size_t)D_ * H_ * 2);
  __bf16* hE_bf     = (__bf16*)alloc((size_t)B_ * H_ * 2);
  __bf16* hEn_bf    = (__bf16*)alloc((size_t)B_ * H_ * 2);
  float*  combined  = (float*)alloc((size_t)B_ * D_ * 4);
  __bf16* comb_bf   = (__bf16*)alloc((size_t)B_ * D_ * 2);
  __bf16* whead_bf  = (__bf16*)alloc((size_t)A_ * D_ * 2);

  dim3 blk(128);
  auto gemmGrid = [](int M, int N) { return dim3(M / BM, (N + BN - 1) / BN); };
  auto cvt = [&](const float* s, __bf16* d, int n) {
    k_cvt<<<dim3((n + 255) / 256), dim3(256), 0, stream>>>(s, d, n);
  };

  // --- input projection: xp = [x, lang] @ W_in^T + b_in (kept in bf16) ---
  int nCat = B_ * (I_ + L_);
  k_concat_bf16<<<dim3((nCat + 255) / 256), dim3(256), 0, stream>>>(x, lang, xcat);
  cvt(W_in, w_in_bf, D_ * (I_ + L_));
  k_gemm_bf16<<<gemmGrid(B_, D_), blk, 0, stream>>>(xcat, w_in_bf, b_in,
      nullptr, xp_bf, B_, D_, I_ + L_);

  // --- router GRU ---
  cvt(Wih_r, wih_r_bf, 3 * HR_ * D_);
  cvt(Whh_r, whh_r_bf, 3 * HR_ * HR_);
  cvt(hR0, hR_bf, B_ * HR_);
  k_gemm_bf16<<<gemmGrid(B_, 3 * HR_), blk, 0, stream>>>(xp_bf, wih_r_bf, bih_r,
      gxbuf, nullptr, B_, 3 * HR_, D_);
  k_gemm_bf16<<<gemmGrid(B_, 3 * HR_), blk, 0, stream>>>(hR_bf, whh_r_bf, bhh_r,
      ghbuf, nullptr, B_, 3 * HR_, HR_);
  k_gru<<<dim3((B_ * HR_ + 255) / 256), dim3(256), 0, stream>>>(
      gxbuf, ghbuf, hR0, out_hR, (__bf16*)nullptr, B_, HR_);
  k_fc_softmax<<<dim3((B_ + 255) / 256), dim3(256), 0, stream>>>(out_hR, W_fc, b_fc, gate);

  // --- experts (sequential; big buffers reused; deterministic accumulate) ---
  k_zero<<<dim3((B_ * D_ + 255) / 256), dim3(256), 0, stream>>>(combined, B_ * D_);
  for (int e = 0; e < E_; ++e) {
    cvt(Wih_e + (size_t)e * 3 * H_ * D_, wih_e_bf, 3 * H_ * D_);
    cvt(Whh_e + (size_t)e * 3 * H_ * H_, whh_e_bf, 3 * H_ * H_);
    cvt(hE0 + (size_t)e * B_ * H_, hE_bf, B_ * H_);
    cvt(W_proj + (size_t)e * D_ * H_, wproj_bf, D_ * H_);
    k_gemm_bf16<<<gemmGrid(B_, 3 * H_), blk, 0, stream>>>(xp_bf, wih_e_bf,
        bih_e + (size_t)e * 3 * H_, gxbuf, nullptr, B_, 3 * H_, D_);
    k_gemm_bf16<<<gemmGrid(B_, 3 * H_), blk, 0, stream>>>(hE_bf, whh_e_bf,
        bhh_e + (size_t)e * 3 * H_, ghbuf, nullptr, B_, 3 * H_, H_);
    k_gru<<<dim3((B_ * H_ + 255) / 256), dim3(256), 0, stream>>>(
        gxbuf, ghbuf, hE0 + (size_t)e * B_ * H_,
        out_hE + (size_t)e * B_ * H_, hEn_bf, B_, H_);
    k_gemm_combine<<<gemmGrid(B_, D_), blk, 0, stream>>>(hEn_bf, wproj_bf,
        b_proj + (size_t)e * D_, gate, e, combined, B_, D_, H_);
  }

  // --- head: logits = combined @ W_head^T + b_head ---
  cvt(combined, comb_bf, B_ * D_);
  cvt(W_head, whead_bf, A_ * D_);
  k_gemm_bf16<<<gemmGrid(B_, A_), blk, 0, stream>>>(comb_bf, whead_bf, b_head,
      out_logits, nullptr, B_, A_, D_);
}
